// FeaturePropagationBlock_33079838113813
// MI455X (gfx1250) — matrix operand order repaired and checked by reading, not compiled, not run on hardware
//
#include <hip/hip_runtime.h>
#include <hip/hip_bf16.h>

// ---------- CDNA5 WMMA types ----------
typedef __bf16 v16bf __attribute__((ext_vector_type(16)));
typedef __bf16 v8bf  __attribute__((ext_vector_type(8)));
typedef float  v8f   __attribute__((ext_vector_type(8)));

// Problem constants (from reference setup_inputs)
#define BB   4
#define NN   16384
#define MM   4096
#define C1C  64
#define C2C  256
#define CIN  320           // C1 + C2
#define H1C  256
#define H2C  128
#define ROWS (BB * NN)     // 65536

__device__ __forceinline__ __bf16 f32_to_bf16(float f) {
    union { float f; unsigned u; } v; v.f = f;
    unsigned r = v.u + 0x7FFFu + ((v.u >> 16) & 1u);   // round-to-nearest-even
    unsigned short h = (unsigned short)(r >> 16);
    __bf16 out; __builtin_memcpy(&out, &h, 2);
    return out;
}

// ---------- CDNA5 async copy: global -> LDS, tracked by ASYNCcnt ----------
__device__ __forceinline__ void async_copy_b128(unsigned lds_byte_addr, const void* gaddr) {
    asm volatile("global_load_async_to_lds_b128 %0, %1, off"
                 :: "v"(lds_byte_addr), "v"(gaddr) : "memory");
}
__device__ __forceinline__ void wait_async0() {
    asm volatile("s_wait_asynccnt 0" ::: "memory");
}
// low 32 bits of a generic (flat) shared pointer == LDS byte offset (ISA 10.2 aperture map)
__device__ __forceinline__ unsigned lds_addr_of(const void* p) {
    return (unsigned)(uintptr_t)p;
}

// ---------- kernel 0: convert weights to bf16, zero stat accumulators ----------
__global__ void prep_kernel(const float* __restrict__ W1, const float* __restrict__ W2,
                            __bf16* __restrict__ W1b, __bf16* __restrict__ W2b,
                            float* __restrict__ stats, int n1, int n2, int nstats) {
    int i = blockIdx.x * blockDim.x + threadIdx.x;
    if (i < n1) W1b[i] = f32_to_bf16(W1[i]);
    if (i < n2) W2b[i] = f32_to_bf16(W2[i]);
    if (i < nstats) stats[i] = 0.0f;
}

// ---------- kernel 1: 3-NN inverse-distance interpolation + concat (bf16 out) ----------
__global__ __launch_bounds__(256)
void knn_interp_kernel(const float* __restrict__ xyz_fine,
                       const float* __restrict__ xyz_coarse,
                       const float* __restrict__ feats_fine,
                       const float* __restrict__ feats_coarse,
                       __bf16* __restrict__ xcat) {
    __shared__ __attribute__((aligned(64))) float sx[MM * 3];   // 48 KB of 320 KB/WGP LDS
    const int b = blockIdx.y;
    const int n = blockIdx.x * blockDim.x + threadIdx.x;

    // async-stage full coarse cloud for this batch (49152 B = 3072 x 16B chunks, uniform)
    {
        const char* gsrc = (const char*)(xyz_coarse + (size_t)b * MM * 3);
        const unsigned lbase = lds_addr_of(&sx[0]);
        for (int i = threadIdx.x; i < (MM * 3 * 4) / 16; i += 256)
            async_copy_b128(lbase + i * 16, gsrc + i * 16);
        wait_async0();
        __syncthreads();
    }

    const float* pf = xyz_fine + ((size_t)b * NN + n) * 3;
    const float px = pf[0], py = pf[1], pz = pf[2];

    float d0 = 3.4e38f, d1 = 3.4e38f, d2 = 3.4e38f;
    int   i0 = 0, i1 = 0, i2 = 0;
    for (int m = 0; m < MM; ++m) {               // LDS broadcast reads
        float dx = px - sx[m * 3 + 0];
        float dy = py - sx[m * 3 + 1];
        float dz = pz - sx[m * 3 + 2];
        float d = dx * dx + dy * dy + dz * dz;
        if (d < d0)      { d2 = d1; i2 = i1; d1 = d0; i1 = i0; d0 = d; i0 = m; }
        else if (d < d1) { d2 = d1; i2 = i1; d1 = d;  i1 = m; }
        else if (d < d2) { d2 = d;  i2 = m; }
    }
    float w0 = 1.0f / fmaxf(d0, 1e-10f);
    float w1 = 1.0f / fmaxf(d1, 1e-10f);
    float w2 = 1.0f / fmaxf(d2, 1e-10f);
    float wsum = 1.0f / (w0 + w1 + w2);
    w0 *= wsum; w1 *= wsum; w2 *= wsum;

    const float* f0 = feats_coarse + ((size_t)b * MM + i0) * C2C;
    const float* f1 = feats_coarse + ((size_t)b * MM + i1) * C2C;
    const float* f2 = feats_coarse + ((size_t)b * MM + i2) * C2C;
    __bf16* xrow = xcat + ((size_t)b * NN + n) * CIN;
#pragma unroll 8
    for (int c = 0; c < C2C; ++c)
        xrow[c] = f32_to_bf16(w0 * f0[c] + w1 * f1[c] + w2 * f2[c]);
    const float* ff = feats_fine + ((size_t)b * NN + n) * C1C;
#pragma unroll 8
    for (int c = 0; c < C1C; ++c)
        xrow[C2C + c] = f32_to_bf16(ff[c]);
}

// ---------- WMMA A-fragment loader (ISA 7.12.2, wave32) ----------
__device__ __forceinline__ v16bf load_a_frag(const __bf16* __restrict__ A, int lda,
                                             int row0, int k0, int lane) {
    const int m = lane & 15;
    const int half = lane >> 4;                  // K-halves across lane groups
    const __bf16* p = A + (size_t)(row0 + m) * lda + k0 + half * 16;
    v8bf lo = *(const v8bf*)p;                   // K = half*16 .. +7
    v8bf hi = *(const v8bf*)(p + 8);             // K = half*16 + 8 .. +7
    v16bf a;
#pragma unroll
    for (int i = 0; i < 8; ++i) { a[i] = lo[i]; a[i + 8] = hi[i]; }
    return a;
}

// ---------- kernel 2: bf16 WMMA GEMM  out[rows,N] = A[rows,K] * W[N,K]^T + bias ----------
// Block = 8 waves = 128 rows x one 64-col strip. The block's 64xK weight slice is
// async-staged to LDS once (ASYNCcnt path) and B-fragments are served by ds_load,
// giving 8-way on-chip reuse; A streams from global/L2.
__global__ __launch_bounds__(256)
void gemm_bf16_kernel(const __bf16* __restrict__ A, const __bf16* __restrict__ W,
                      const float* __restrict__ bias, float* __restrict__ out,
                      int rows, int N, int K) {
    __shared__ __attribute__((aligned(64))) __bf16 wlds[64 * CIN];  // max 40 KB
    const int lane = threadIdx.x & 31;
    const int wave = threadIdx.x >> 5;
    const int strips = N >> 6;                               // N/64 column strips
    const int n0   = (blockIdx.x % strips) * 64;
    const int row0 = (blockIdx.x / strips) * 128 + wave * 16;

    // ---- async-stage W rows [n0, n0+64) x K (contiguous in row-major W[N,K]) ----
    {
        const char* gsrc = (const char*)(W + (size_t)n0 * K);
        const unsigned lbase = lds_addr_of(&wlds[0]);
        const int chunks = (64 * K * 2) / 16;                // 2560 (K=320) / 2048 (K=256)
        for (int i = threadIdx.x; i < chunks; i += 256)      // uniform trip count
            async_copy_b128(lbase + i * 16, gsrc + i * 16);
        wait_async0();                                       // this wave's copies done
        __syncthreads();                                     // all waves' copies done
    }

    v8f acc[4] = {};
    const int nb = lane & 15;
    const int kb = (lane >> 4) * 16;                         // B frag: lanes 0-15 K 0-15, 16-31 K 16-31
    for (int k0 = 0; k0 < K; k0 += 32) {
        v16bf a = load_a_frag(A, K, row0, k0, lane);         // reused by 4 WMMAs
#pragma unroll
        for (int s = 0; s < 4; ++s) {
            v16bf bfrag = *(const v16bf*)&wlds[(s * 16 + nb) * K + k0 + kb];  // ds_load
            acc[s] = __builtin_amdgcn_wmma_f32_16x16x32_bf16(
                false, a, false, bfrag, (short)0, acc[s], false, false);
        }
    }

    const int mh = (lane >> 4) * 8;                          // C/D: VGPR j -> row j + 8*hi
#pragma unroll
    for (int s = 0; s < 4; ++s) {
        const int ncol = n0 + s * 16 + nb;
        const float bv = bias[ncol];
        float* orow = out + (size_t)(row0 + mh) * N + ncol;
#pragma unroll
        for (int j = 0; j < 8; ++j)
            orow[(size_t)j * N] = acc[s][j] + bv;
    }
}

// ---------- kernel 3: per-channel sum / sumsq (coalesced column reduction) ----------
__global__ void bn_stats_kernel(const float* __restrict__ y, float* __restrict__ sum,
                                float* __restrict__ sumsq, int rows, int N,
                                int rows_per_block) {
    const int c = threadIdx.x;                   // blockDim.x == N (128 or 256)
    const int r0 = blockIdx.x * rows_per_block;
    int r1 = r0 + rows_per_block; if (r1 > rows) r1 = rows;
    float s = 0.0f, q = 0.0f;
    for (int r = r0; r < r1; ++r) {
        float v = y[(size_t)r * N + c];          // lanes read consecutive channels
        s += v; q += v * v;
    }
    atomicAdd(&sum[c], s);
    atomicAdd(&sumsq[c], q);
}

// ---------- kernel 4: BatchNorm + ReLU (f32 in-place, or bf16 for next GEMM) ----------
__global__ void bn_apply_kernel(const float* __restrict__ y,
                                const float* __restrict__ sum, const float* __restrict__ sumsq,
                                const float* __restrict__ gamma, const float* __restrict__ beta,
                                float* __restrict__ out_f32, __bf16* __restrict__ out_bf16,
                                int N, size_t total, float inv_cnt) {
    size_t idx = (size_t)blockIdx.x * blockDim.x + threadIdx.x;
    if (idx >= total) return;
    const int c = (int)(idx & (size_t)(N - 1));  // N is a power of two
    const float mean = sum[c] * inv_cnt;
    const float var  = sumsq[c] * inv_cnt - mean * mean;
    const float sc   = gamma[c] * rsqrtf(var + 1e-5f);
    float v = (y[idx] - mean) * sc + beta[c];
    v = fmaxf(v, 0.0f);
    if (out_f32) out_f32[idx] = v;
    else         out_bf16[idx] = f32_to_bf16(v);
}

extern "C" void kernel_launch(void* const* d_in, const int* in_sizes, int n_in,
                              void* d_out, int out_size, void* d_ws, size_t ws_size,
                              hipStream_t stream) {
    (void)in_sizes; (void)n_in; (void)out_size; (void)ws_size;
    const float* xyz_fine     = (const float*)d_in[0];
    const float* xyz_coarse   = (const float*)d_in[1];
    const float* feats_fine   = (const float*)d_in[2];
    const float* feats_coarse = (const float*)d_in[3];
    const float* W1  = (const float*)d_in[4];
    const float* b1  = (const float*)d_in[5];
    const float* g1  = (const float*)d_in[6];
    const float* be1 = (const float*)d_in[7];
    const float* W2  = (const float*)d_in[8];
    const float* b2  = (const float*)d_in[9];
    const float* g2  = (const float*)d_in[10];
    const float* be2 = (const float*)d_in[11];
    float* out = (float*)d_out;

    // ---- workspace carve-up (all 32B-aligned) ----
    __bf16* x1  = (__bf16*)d_ws;                         // [ROWS, CIN]  bf16
    __bf16* w1b = x1  + (size_t)ROWS * CIN;              // [H1, CIN]    bf16
    __bf16* w2b = w1b + (size_t)H1C * CIN;               // [H2, H1]     bf16
    float*  y1  = (float*)(w2b + (size_t)H2C * H1C);     // [ROWS, H1]   f32
    __bf16* x2  = (__bf16*)(y1 + (size_t)ROWS * H1C);    // [ROWS, H1]   bf16
    float*  st  = (float*)(x2 + (size_t)ROWS * H1C);     // stats: 768 floats
    float *sum1 = st, *sq1 = st + 256, *sum2 = st + 512, *sq2 = st + 640;

    const float inv_cnt = 1.0f / (float)ROWS;

    // 0) weights -> bf16, zero stats
    prep_kernel<<<(H1C * CIN + 255) / 256, 256, 0, stream>>>(
        W1, W2, w1b, w2b, st, H1C * CIN, H2C * H1C, 768);

    // 1) kNN-3 + interpolate + concat -> x1 (bf16)
    dim3 kg(NN / 256, BB);
    knn_interp_kernel<<<kg, 256, 0, stream>>>(
        xyz_fine, xyz_coarse, feats_fine, feats_coarse, x1);

    // 2) layer 1: GEMM (ROWS x 320) * (320 x 256) -> y1
    gemm_bf16_kernel<<<(ROWS / 128) * (H1C / 64), 256, 0, stream>>>(
        x1, w1b, b1, y1, ROWS, H1C, CIN);
    bn_stats_kernel<<<256, H1C, 0, stream>>>(y1, sum1, sq1, ROWS, H1C, ROWS / 256);
    bn_apply_kernel<<<(int)(((size_t)ROWS * H1C + 255) / 256), 256, 0, stream>>>(
        y1, sum1, sq1, g1, be1, nullptr, x2, H1C, (size_t)ROWS * H1C, inv_cnt);

    // 3) layer 2: GEMM (ROWS x 256) * (256 x 128) -> d_out
    gemm_bf16_kernel<<<(ROWS / 128) * (H2C / 64), 256, 0, stream>>>(
        x2, w2b, b2, out, ROWS, H2C, H1C);
    bn_stats_kernel<<<256, H2C, 0, stream>>>(out, sum2, sq2, ROWS, H2C, ROWS / 256);
    bn_apply_kernel<<<(int)(((size_t)ROWS * H2C + 255) / 256), 256, 0, stream>>>(
        out, sum2, sq2, g2, be2, out, nullptr, H2C, (size_t)ROWS * H2C, inv_cnt);
}
